// MoDE_2087354106147
// MI455X (gfx1250) — matrix operand order repaired and checked by reading, not compiled, not run on hardware
//
#include <hip/hip_runtime.h>
#include <cmath>

#define DEV __device__ __forceinline__

typedef __attribute__((ext_vector_type(8)))  float    v8f;
typedef __attribute__((ext_vector_type(16))) __bf16   v16bf;
typedef __attribute__((ext_vector_type(4)))  unsigned u32x4;
typedef __attribute__((ext_vector_type(4)))  int      i32x4;
typedef __attribute__((ext_vector_type(8)))  int      i32x8;

// LDS pixel-row stride: 64 data ch + 8 pad elements -> 144B, bank-conflict-free
#define LSTR 72

// A/B fragment for v_wmma_f32_16x16x32_bf16 (16 bf16 = 8 dwords per lane)
union Frag {
    v16bf          v;
    unsigned short h[16];
    unsigned       u[8];
    u32x4          q[2];
};

DEV unsigned short f2bf(float f) {
    unsigned u = __float_as_uint(f);
    u += 0x7FFFu + ((u >> 16) & 1u);           // round-to-nearest-even
    return (unsigned short)(u >> 16);
}
DEV float bf2f(unsigned short h) { return __uint_as_float(((unsigned)h) << 16); }

DEV v8f wmma_bf16(const Frag& a, const Frag& b, v8f c) {
    return __builtin_amdgcn_wmma_f32_16x16x32_bf16(false, a.v, false, b.v,
                                                   (short)0, c, false, false);
}

// ---- Tensor Data Mover: load bf16 tile (contiguous [d2][d1][d0]) into LDS ----
// D# packing per CDNA5 ISA.
// LDS padding: after every 32 DWORDs (one 64-ch row) skip 4 DWORDs (8 elems)
// -> per-pixel LDS stride of 72 elements (conflict-free B-fragment reads).
DEV void tdm_load_bf16(unsigned ldsAddr, const void* gaddr,
                       unsigned tileD0, unsigned tileD1, unsigned tileD2,
                       unsigned tenD0, unsigned tenD1, unsigned tenD2,
                       unsigned long long s0, unsigned long long s1)
{
    unsigned long long ga = (unsigned long long)gaddr;
    u32x4 g0;
    g0[0] = 1u;                                              // count=1, user mode
    g0[1] = ldsAddr;                                         // lds_addr
    g0[2] = (unsigned)ga;                                    // global_addr lo
    g0[3] = ((unsigned)(ga >> 32) & 0x1FFFFFFu) | 0x80000000u; // addr hi | type=2
    i32x8 g1;
    g1[0] = (int)((1u << 16)                                 // data_size=1 (2 bytes)
                | (1u << 20)                                 // pad_enable
                | (4u << 22)                                 // pad_interval: 32 DWORDs
                | (3u << 25));                               // pad_amount: 4 DWORDs
    g1[1] = (int)((tenD0 & 0xFFFFu) << 16);                  // tensor_dim0[15:0]
    g1[2] = (int)((tenD0 >> 16) | ((tenD1 & 0xFFFFu) << 16));
    g1[3] = (int)((tenD1 >> 16) | (tileD0 << 16));           // tile_dim0
    g1[4] = (int)(tileD1 | (tileD2 << 16));                  // tile_dim1/2
    g1[5] = (int)(unsigned)(s0 & 0xFFFFFFFFull);             // tensor_dim0_stride
    g1[6] = (int)((unsigned)(s0 >> 32) | (((unsigned)s1 & 0xFFFFu) << 16));
    g1[7] = (int)(unsigned)(s1 >> 16);                       // tensor_dim1_stride hi
    i32x4 g2 = { (int)tenD2, 0, 0, 0 };                      // tensor_dim2 (3D)
    i32x4 g3 = { 0, 0, 0, 0 };
#if defined(__clang_major__) && (__clang_major__ >= 23)
    i32x8 g4 = { 0, 0, 0, 0, 0, 0, 0, 0 };
    __builtin_amdgcn_tensor_load_to_lds(g0, g1, g2, g3, g4, 0);
#else
    __builtin_amdgcn_tensor_load_to_lds(g0, g1, g2, g3, 0);
#endif
}

// GELU(exact) + pack 8 consecutive channels -> one ds_store_b128 per M-tile
DEV void gelu_store(unsigned short* sHp, v8f acc[3], int hp, int hi, bool inImg) {
#pragma unroll
    for (int mt = 0; mt < 3; ++mt) {
        unsigned short pk[8];
#pragma unroll
        for (int r = 0; r < 8; ++r) {
            const float vv = acc[mt][r];
            const float gl = 0.5f * vv * (1.f + erff(vv * 0.70710678118f));
            pk[r] = inImg ? f2bf(gl) : (unsigned short)0;
        }
        *(u32x4*)&sHp[hp * LSTR + mt * 16 + 8 * hi] = *(const u32x4*)pk;
    }
}

// =====================  prep kernels  =====================

// x [b][c][pix] f32  ->  xbf64 [b][pix][64] bf16 (channels 48..63 zero)
__global__ __launch_bounds__(256) void k_x_pack(const float* __restrict__ x,
                                                unsigned short* __restrict__ xbf64) {
    const int t = threadIdx.x;
    const int pix = (blockIdx.x & 255) * 256 + t;
    const int b = blockIdx.x >> 8;
    unsigned short row[64];
#pragma unroll
    for (int c = 0; c < 48; ++c)
        row[c] = f2bf(x[(((size_t)(b * 48 + c)) << 16) + pix]);
#pragma unroll
    for (int j = 48; j < 64; ++j) row[j] = 0;
    u32x4* dst = (u32x4*)(xbf64 + ((size_t)(b * 65536 + pix)) * 64);
    const u32x4* src = (const u32x4*)row;
#pragma unroll
    for (int q = 0; q < 8; ++q) dst[q] = src[q];
}

// proj weights -> [oc(48)][ic(64, zero-padded)] bf16
__global__ void k_pack_proj(const float* __restrict__ wa, const float* __restrict__ wb,
                            unsigned short* __restrict__ pa, unsigned short* __restrict__ pb) {
    int i = blockIdx.x * blockDim.x + threadIdx.x;
    if (i >= 48 * 64) return;
    int ic = i & 63, oc = i >> 6;
    pa[i] = (ic < 48) ? f2bf(wa[oc * 48 + ic]) : (unsigned short)0;
    pb[i] = (ic < 48) ? f2bf(wb[oc * 48 + ic]) : (unsigned short)0;
}

// expert weights -> [e][conv(2)][tap(9)][oc(48)][ic(64 padded)] bf16
__global__ void k_pack_experts(const float* __restrict__ w1, const float* __restrict__ w2,
                               unsigned short* __restrict__ dst) {
    int i = blockIdx.x * blockDim.x + threadIdx.x;
    if (i >= 8 * 2 * 9 * 48 * 64) return;
    int ic = i & 63; int t = i >> 6;
    int oc = t % 48; t /= 48;
    int tap = t % 9; t /= 9;
    int cv = t & 1;  int e = t >> 1;
    unsigned short v = 0;
    if (ic < 48) {
        int kh = tap / 3, kw = tap % 3;
        const float* src = cv ? w2 : w1;
        v = f2bf(src[(((e * 48 + oc) * 48 + ic) * 3 + kh) * 3 + kw]);
    }
    dst[i] = v;
}

__global__ void k_prompt_mean_and_zero(const float* __restrict__ P,
                                       float* __restrict__ pavg, float* __restrict__ gap) {
    int i = threadIdx.x;                 // 384 = 8*48
    if (i < 384) {
        int b = i / 48, c = i % 48;
        float s = 0.f;
#pragma unroll
        for (int p = 0; p < 16; ++p) s += P[(b * 16 + p) * 48 + c];
        pavg[i] = s * (1.f / 16.f);
        gap[i] = 0.f;
    }
}

// =====================  branch a/b projections + prompt einsum  =====================
// per block: one (b, y) image row. TDM loads [256 px][64 ch] with 72-stride LDS padding.
__global__ __launch_bounds__(256) void k_proj_fi(
    const unsigned short* __restrict__ xbf64, const float* __restrict__ pavg,
    const u32x4* __restrict__ pkA, const u32x4* __restrict__ pkB,
    unsigned short* __restrict__ xbp, float* __restrict__ fi4)
{
    __shared__ __align__(16) unsigned short xrow[256 * LSTR];   // [pix][72]
    __shared__ float pv[48];
    const int tid = threadIdx.x;
    const int lane = tid & 31, wv = tid >> 5;
    const int hi = (lane >> 4) & 1, l15 = lane & 15;
    const int y = blockIdx.x & 255, b = blockIdx.x >> 8;

    if (tid < 48) pv[tid] = pavg[b * 48 + tid];
    if (wv == 0) {
        unsigned lds = (unsigned)(unsigned long long)(void*)&xrow[0];
        const void* ga = xbf64 + ((size_t)(b * 65536 + y * 256)) * 64;
        // 2D tile: [256 px][64 ch], pixel stride = 64 elements in memory
        tdm_load_bf16(lds, ga, 64, 256, 0, 64, 65536, 0, 64ull, 0ull);
        __builtin_amdgcn_s_wait_tensorcnt(0);
    }
    __syncthreads();

    const v8f vz = {0.f,0.f,0.f,0.f,0.f,0.f,0.f,0.f};
    for (int nt = 0; nt < 2; ++nt) {
        const int px = wv * 32 + nt * 16 + l15;
        v8f aA[3] = { vz, vz, vz };
        v8f aB[3] = { vz, vz, vz };
#pragma unroll
        for (int ck = 0; ck < 64; ck += 32) {
            const u32x4* bp = (const u32x4*)&xrow[px * LSTR + ck + hi * 8];
            Frag bf; bf.q[0] = bp[0]; bf.q[1] = bp[2];       // 2 x ds_load_b128
#pragma unroll
            for (int mt = 0; mt < 3; ++mt) {
                const int qi = (mt * 16 + l15) * 8 + (ck >> 3) + hi;
                Frag af;  af.q[0] = pkA[qi]; af.q[1] = pkA[qi + 2];
                aA[mt] = wmma_bf16(af, bf, aA[mt]);
                Frag af2; af2.q[0] = pkB[qi]; af2.q[1] = pkB[qi + 2];
                aB[mt] = wmma_bf16(af2, bf, aB[mt]);
            }
        }
        // fi[b,g,pix] = sum_{c in group g} p_avg[c] * xa[c,pix]
        float f0 = 0.f, f1 = 0.f, fg2 = 0.f, f3 = 0.f;
#pragma unroll
        for (int mt = 0; mt < 3; ++mt)
#pragma unroll
            for (int r = 0; r < 8; ++r) {
                const int M = mt * 16 + r + 8 * hi;
                const float ct = pv[M] * aA[mt][r];
                const int g = M / 12;
                f0  += (g == 0) ? ct : 0.f;
                f1  += (g == 1) ? ct : 0.f;
                fg2 += (g == 2) ? ct : 0.f;
                f3  += (g == 3) ? ct : 0.f;
            }
        f0  += __shfl_xor(f0, 16, 32);
        f1  += __shfl_xor(f1, 16, 32);
        fg2 += __shfl_xor(fg2, 16, 32);
        f3  += __shfl_xor(f3, 16, 32);
        if (lane < 16) {
            const int pix = y * 256 + px;
            fi4[((b * 4 + 0) << 16) + pix] = f0;
            fi4[((b * 4 + 1) << 16) + pix] = f1;
            fi4[((b * 4 + 2) << 16) + pix] = fg2;
            fi4[((b * 4 + 3) << 16) + pix] = f3;
        }
#pragma unroll
        for (int mt = 0; mt < 3; ++mt)
#pragma unroll
            for (int r = 0; r < 8; ++r) {
                const int c = mt * 16 + r + 8 * hi;
                xbp[((b * 48 + c) << 16) + y * 256 + px] = f2bf(aB[mt][r]);
            }
    }
}

// =====================  Fx = dw3x3(xb) + fi_align(fi4)  (channel-last 64-padded) + GAP  =====================
__global__ __launch_bounds__(256) void k_fx_gap(
    const unsigned short* __restrict__ xbp, const float* __restrict__ fi4,
    const float* __restrict__ dww, const float* __restrict__ fiw,
    unsigned short* __restrict__ fxbf64, float* __restrict__ gap)
{
    __shared__ float sgap[48];
    __shared__ float sdw[432];   // dw weights [48][9]
    __shared__ float sfw[192];   // fi_align weights [48][4]
    const int tid = threadIdx.x;
    const int y = blockIdx.x & 255, b = blockIdx.x >> 8;
    const int x = tid;
    if (tid < 48) sgap[tid] = 0.f;
    for (int i = tid; i < 432; i += 256) sdw[i] = dww[i];
    if (tid < 192) sfw[tid] = fiw[tid];
    __syncthreads();

    float fi[4];
#pragma unroll
    for (int g = 0; g < 4; ++g) fi[g] = fi4[((b * 4 + g) << 16) + y * 256 + x];

    unsigned short row[64];
#pragma unroll
    for (int j = 48; j < 64; ++j) row[j] = 0;

#pragma unroll
    for (int c = 0; c < 48; ++c) {
        float acc = 0.f;
        const unsigned short* xc = xbp + (((size_t)(b * 48 + c)) << 16);
#pragma unroll
        for (int kh = 0; kh < 3; ++kh) {
            const int yy = y + kh - 1;
            if (yy >= 0 && yy < 256) {
#pragma unroll
                for (int kw = 0; kw < 3; ++kw) {
                    const int xx = x + kw - 1;
                    if (xx >= 0 && xx < 256)
                        acc += sdw[c * 9 + kh * 3 + kw] * bf2f(xc[yy * 256 + xx]);
                }
            }
        }
#pragma unroll
        for (int g = 0; g < 4; ++g) acc += sfw[c * 4 + g] * fi[g];
        row[c] = f2bf(acc);
        float s = acc;                                  // GAP partial
#pragma unroll
        for (int o = 16; o > 0; o >>= 1) s += __shfl_xor(s, o, 32);
        if ((tid & 31) == 0) atomicAdd(&sgap[c], s);    // ds_add_f32
    }
    u32x4* dst = (u32x4*)(fxbf64 + ((size_t)(b * 65536 + y * 256 + x)) * 64);
    const u32x4* src = (const u32x4*)row;
#pragma unroll
    for (int q = 0; q < 8; ++q) dst[q] = src[q];        // coalesced b128 row store

    __syncthreads();
    if (tid < 48) atomicAdd(&gap[b * 48 + tid], sgap[tid]);
}

// =====================  router: exact top-2 softmax gate  =====================
__global__ void k_router(const float* __restrict__ gap, const float* __restrict__ rw,
                         const float* __restrict__ rb, float* __restrict__ gate)
{
    __shared__ float sc[64];
    const int tid = threadIdx.x;
    if (tid < 64) {
        const int b = tid >> 3, e = tid & 7;
        float s = rb[e];
        for (int c = 0; c < 48; ++c)
            s += (gap[b * 48 + c] * (1.f / 65536.f)) * rw[e * 48 + c];
        sc[tid] = s;
    }
    __syncthreads();
    if (tid < 8) {
        const int b = tid;
        const float* s = &sc[b * 8];
        int i1 = 0; float v1 = s[0];
        for (int e = 1; e < 8; ++e) if (s[e] > v1) { v1 = s[e]; i1 = e; }
        int i2 = -1; float v2 = -3.4e38f;
        for (int e = 0; e < 8; ++e) if (e != i1 && s[e] > v2) { v2 = s[e]; i2 = e; }
        const float e2 = expf(v2 - v1);
        const float inv = 1.f / (1.f + e2);
        for (int e = 0; e < 8; ++e) gate[b * 8 + e] = 0.f;
        gate[b * 8 + i1] = inv;
        gate[b * 8 + i2] = e2 * inv;
    }
}

// =====================  gated experts: conv3x3 -> gelu -> conv3x3, + residual  =====================
// block = 16x8 output tile. Fx halo tile [12 y][20 x][64 ch] via TDM (border: guarded loads).
__global__ __launch_bounds__(256) void k_experts(
    const unsigned short* __restrict__ fxbf64, const float* __restrict__ xin,
    const float* __restrict__ gate, const u32x4* __restrict__ wpk,
    float* __restrict__ out)
{
    __shared__ __align__(16) unsigned short sFx[240 * LSTR];   // [py*20+px][72]
    __shared__ __align__(16) unsigned short sH[192 * LSTR];    // [hy*18+hx][72]
    const int tid = threadIdx.x;
    const int lane = tid & 31, wv = tid >> 5;
    const int hi = (lane >> 4) & 1, l15 = lane & 15;
    const int tx = blockIdx.x & 15;
    const int ty = (blockIdx.x >> 4) & 31;
    const int b  = blockIdx.x >> 9;
    const int y0 = ty * 8 - 2, x0 = tx * 16 - 2;

    // zero pad channels 48..63 of sH (conv1 only writes oc<48)
    {
        const u32x4 z = {0, 0, 0, 0};
        for (int i = tid; i < 192 * 2; i += 256)
            *(u32x4*)&sH[(i >> 1) * LSTR + 48 + (i & 1) * 8] = z;
    }

    const bool interior = (y0 >= 0) && (x0 >= 0) && (y0 + 12 <= 256) && (x0 + 20 <= 256);
    if (interior) {
        if (wv == 0) {
            unsigned lds = (unsigned)(unsigned long long)(void*)&sFx[0];
            const void* ga = fxbf64 + ((size_t)(b * 65536 + y0 * 256 + x0)) * 64;
            // 3D tile [12 y][20 x][64 ch] of tensor [256][256][64]
            tdm_load_bf16(lds, ga, 64, 20, 12, 64, 256, 256, 64ull, 16384ull);
            __builtin_amdgcn_s_wait_tensorcnt(0);
        }
    } else {
        for (int p = tid; p < 240; p += 256) {
            const int py = p / 20, pxx = p % 20;
            const int gy = y0 + py, gx = x0 + pxx;
            u32x4* dst = (u32x4*)&sFx[p * LSTR];
            if (gy >= 0 && gy < 256 && gx >= 0 && gx < 256) {
                const u32x4* src = (const u32x4*)(fxbf64 + ((size_t)(b * 65536 + gy * 256 + gx)) * 64);
#pragma unroll
                for (int q = 0; q < 8; ++q) dst[q] = src[q];
            } else {
                const u32x4 z = {0, 0, 0, 0};
#pragma unroll
                for (int q = 0; q < 8; ++q) dst[q] = z;
            }
        }
    }
    __syncthreads();

    const v8f vz = {0.f,0.f,0.f,0.f,0.f,0.f,0.f,0.f};
    v8f oacc[3] = { vz, vz, vz };

    for (int e = 0; e < 8; ++e) {
        const float ge = gate[b * 8 + e];
        if (ge == 0.f) continue;          // uniform across the block
        // hint the gated expert's weight block into cache (global_prefetch_b8)
        __builtin_prefetch(((const char*)wpk) + (size_t)(e * 2) * 9 * 48 * 64 * 2, 0, 1);
        __syncthreads();                  // protect sH from previous conv2 readers

        // ---- conv1: two resident pixel-tiles per wave, A-fragments shared ----
        {
            const int t1 = wv + 8;
            const bool has2 = (t1 < 12);
            int hp0 = wv * 16 + l15;            if (hp0 > 179) hp0 = 179;
            int hp1 = has2 ? t1 * 16 + l15 : hp0; if (hp1 > 179) hp1 = 179;
            const int hy0 = hp0 / 18, hx0 = hp0 % 18;
            const int hy1 = hp1 / 18, hx1 = hp1 % 18;
            v8f acc0[3] = { vz, vz, vz };
            v8f acc1[3] = { vz, vz, vz };
            for (int tap = 0; tap < 9; ++tap) {
                const int kh = tap / 3, kw = tap % 3;
                const int po0 = (hy0 + kh) * 20 + (hx0 + kw);
                const int po1 = (hy1 + kh) * 20 + (hx1 + kw);
                const int rb0 = ((e * 2 + 0) * 9 + tap) * 48;
#pragma unroll
                for (int ck = 0; ck < 64; ck += 32) {
                    Frag af[3];
#pragma unroll
                    for (int mt = 0; mt < 3; ++mt) {
                        const int qi = (rb0 + mt * 16 + l15) * 8 + (ck >> 3) + hi;
                        af[mt].q[0] = wpk[qi]; af[mt].q[1] = wpk[qi + 2];
                    }
                    const u32x4* bp0 = (const u32x4*)&sFx[po0 * LSTR + ck + hi * 8];
                    Frag b0; b0.q[0] = bp0[0]; b0.q[1] = bp0[2];
#pragma unroll
                    for (int mt = 0; mt < 3; ++mt) acc0[mt] = wmma_bf16(af[mt], b0, acc0[mt]);
                    if (has2) {
                        const u32x4* bp1 = (const u32x4*)&sFx[po1 * LSTR + ck + hi * 8];
                        Frag b1; b1.q[0] = bp1[0]; b1.q[1] = bp1[2];
#pragma unroll
                        for (int mt = 0; mt < 3; ++mt) acc1[mt] = wmma_bf16(af[mt], b1, acc1[mt]);
                    }
                }
            }
            {
                const int gy = y0 + 1 + hy0, gx = x0 + 1 + hx0;
                gelu_store(sH, acc0, hp0, hi,
                           (gy >= 0) && (gy < 256) && (gx >= 0) && (gx < 256));
            }
            if (has2) {
                const int gy = y0 + 1 + hy1, gx = x0 + 1 + hx1;
                gelu_store(sH, acc1, hp1, hi,
                           (gy >= 0) && (gy < 256) && (gx >= 0) && (gx < 256));
            }
        }
        __syncthreads();
        // ---- conv2 (implicit GEMM) -> gate-weighted accumulate ----
        {
            v8f acc[3] = { vz, vz, vz };
            for (int tap = 0; tap < 9; ++tap) {
                const int kh = tap / 3, kw = tap % 3;
                const int pixoff = (wv + kh) * 18 + (l15 + kw);
                const int rb1 = ((e * 2 + 1) * 9 + tap) * 48;
#pragma unroll
                for (int ck = 0; ck < 64; ck += 32) {
                    const u32x4* bp = (const u32x4*)&sH[pixoff * LSTR + ck + hi * 8];
                    Frag bf; bf.q[0] = bp[0]; bf.q[1] = bp[2];
#pragma unroll
                    for (int mt = 0; mt < 3; ++mt) {
                        const int qi = (rb1 + mt * 16 + l15) * 8 + (ck >> 3) + hi;
                        Frag af; af.q[0] = wpk[qi]; af.q[1] = wpk[qi + 2];
                        acc[mt] = wmma_bf16(af, bf, acc[mt]);
                    }
                }
            }
#pragma unroll
            for (int mt = 0; mt < 3; ++mt)
#pragma unroll
                for (int r = 0; r < 8; ++r) oacc[mt][r] += ge * acc[mt][r];
        }
    }
    // ---- residual add + store ----
    const int gy = ty * 8 + wv;
    const int gxb = tx * 16 + l15;
#pragma unroll
    for (int mt = 0; mt < 3; ++mt)
#pragma unroll
        for (int r = 0; r < 8; ++r) {
            const int c = mt * 16 + r + 8 * hi;
            const int idx = ((b * 48 + c) << 16) + gy * 256 + gxb;
            out[idx] = oacc[mt][r] + xin[idx];
        }
}

// =====================  host launcher  =====================

extern "C" void kernel_launch(void* const* d_in, const int* in_sizes, int n_in,
                              void* d_out, int out_size, void* d_ws, size_t ws_size,
                              hipStream_t stream)
{
    (void)in_sizes; (void)n_in; (void)out_size; (void)ws_size;
    const float* x        = (const float*)d_in[0];
    const float* P_hat    = (const float*)d_in[1];
    const float* proj_a_w = (const float*)d_in[2];
    const float* proj_b_w = (const float*)d_in[3];
    const float* dw_b_w   = (const float*)d_in[4];
    const float* fi_align = (const float*)d_in[5];
    const float* exp_w1   = (const float*)d_in[6];
    const float* exp_w2   = (const float*)d_in[7];
    const float* router_w = (const float*)d_in[8];
    const float* router_b = (const float*)d_in[9];
    float* out = (float*)d_out;

    char* ws = (char*)d_ws;
    size_t off = 0;
    auto alloc = [&](size_t bytes) -> char* {
        off = (off + 255) & ~(size_t)255;
        char* p = ws + off; off += bytes; return p;
    };
    // xbf64 is dead after k_proj_fi; reuse the same buffer for fxbf64.
    unsigned short* xbf64  = (unsigned short*)alloc(8ull * 65536 * 64 * 2);  // 67 MB
    unsigned short* fxbf64 = xbf64;
    unsigned short* xbp  = (unsigned short*)alloc(25165824ull * 2);          // 50 MB
    float* fi4  = (float*)alloc(8ull * 4 * 65536 * 4);
    float* gap  = (float*)alloc(384 * 4);
    float* pavg = (float*)alloc(384 * 4);
    float* gate = (float*)alloc(64 * 4);
    unsigned short* pkA = (unsigned short*)alloc(48 * 64 * 2);
    unsigned short* pkB = (unsigned short*)alloc(48 * 64 * 2);
    unsigned short* wpk = (unsigned short*)alloc(8ull * 2 * 9 * 48 * 64 * 2);

    k_x_pack<<<8 * 256, 256, 0, stream>>>(x, xbf64);
    k_pack_proj<<<(48 * 64 + 255) / 256, 256, 0, stream>>>(proj_a_w, proj_b_w, pkA, pkB);
    k_pack_experts<<<(8 * 2 * 9 * 48 * 64 + 255) / 256, 256, 0, stream>>>(exp_w1, exp_w2, wpk);
    k_prompt_mean_and_zero<<<1, 384, 0, stream>>>(P_hat, pavg, gap);
    k_proj_fi<<<8 * 256, 256, 0, stream>>>(xbf64, pavg, (const u32x4*)pkA, (const u32x4*)pkB, xbp, fi4);
    k_fx_gap<<<8 * 256, 256, 0, stream>>>(xbp, fi4, dw_b_w, fi_align, fxbf64, gap);
    k_router<<<1, 64, 0, stream>>>(gap, router_w, router_b, gate);
    k_experts<<<8 * 32 * 16, 256, 0, stream>>>(fxbf64, x, gate, (const u32x4*)wpk, out);
}
